// BioAwareGNN_49813030699072
// MI455X (gfx1250) — matrix operand order repaired and checked by reading, not compiled, not run on hardware
//
#include <hip/hip_runtime.h>
#include <hip/hip_bf16.h>

// ---------------- problem constants ----------------
#define N_NODES 16384
#define N_EDGES 131072
#define BGR     64
#define NPG     256
#define INF_D   32
#define NH      4
#define CC      96
#define HC      384
#define SEQCH   48
#define EMBD    12
#define EPS_BN  1e-5f

typedef __attribute__((ext_vector_type(16))) __bf16 v16bf;
typedef __attribute__((ext_vector_type(8)))  float  v8f;
typedef __attribute__((ext_vector_type(4)))  float  f4;

// ---------------- fragment builders (all contiguous, vectorized) ----------------
// A (16x32 bf16) per-lane layout: element i -> K = half*16 + i (contiguous 16)
// B (32x16 bf16) per-lane layout: element i -> K = half*16 + i (contiguous 16)
__device__ __forceinline__ void pack8(v16bf& dst, int base, const float* p) {
  f4 a = *(const f4*)p;
  f4 b = *(const f4*)(p + 4);
  dst[base + 0] = (__bf16)a[0]; dst[base + 1] = (__bf16)a[1];
  dst[base + 2] = (__bf16)a[2]; dst[base + 3] = (__bf16)a[3];
  dst[base + 4] = (__bf16)b[0]; dst[base + 5] = (__bf16)b[1];
  dst[base + 6] = (__bf16)b[2]; dst[base + 7] = (__bf16)b[3];
}
__device__ __forceinline__ v16bf make_a_frag(const float* p, int half) {
  v16bf a;
  pack8(a, 0, p + half * 16);
  pack8(a, 8, p + half * 16 + 8);
  return a;
}
__device__ __forceinline__ v16bf make_b_frag_f32(const float* p, int half) {
  v16bf b;
  pack8(b, 0, p + half * 16);
  pack8(b, 8, p + half * 16 + 8);
  return b;
}

__device__ __forceinline__ unsigned f2ord(float f) {
  unsigned u = __float_as_uint(f);
  return (u & 0x80000000u) ? ~u : (u | 0x80000000u);
}
__device__ __forceinline__ float ord2f(unsigned u) {
  return (u & 0x80000000u) ? __uint_as_float(u & 0x7FFFFFFFu)
                           : __uint_as_float(~u);
}

// =====================================================================
// Weight preparation kernels (run once per call; tiny)
// =====================================================================
// W[K,N] f32 (row-major)  ->  Wt[N,K] bf16   (for jax-style y = x @ W)
__global__ void k_w_nn_to_bf_t(const float* __restrict__ W, __bf16* __restrict__ Wt,
                               int K, int Nn) {
  int i = blockIdx.x * blockDim.x + threadIdx.x;
  if (i >= K * Nn) return;
  int n = i / K, k = i % K;
  Wt[i] = (__bf16)W[(size_t)k * Nn + n];
}
// straight f32 -> bf16 copy (for torch-style W[N,K], y = x @ W^T)
__global__ void k_f32_to_bf(const float* __restrict__ W, __bf16* __restrict__ O, int n) {
  int i = blockIdx.x * blockDim.x + threadIdx.x;
  if (i < n) O[i] = (__bf16)W[i];
}
// Vt[bh, c(96), j(256)] bf16 from qkv (V slice), so P@V has contiguous B-side
__global__ void k_make_vt(const float* __restrict__ qkv, __bf16* __restrict__ Vt) {
  int i = blockIdx.x * blockDim.x + threadIdx.x;
  if (i >= BGR * NH * CC * NPG) return;
  int j = i & 255;
  int rest = i >> 8;
  int c = rest % CC;
  int bh = rest / CC;
  int b = bh >> 2, h = bh & 3;
  Vt[i] = (__bf16)qkv[(size_t)(b * NPG + j) * (3 * HC) + 2 * HC + h * CC + c];
}

// =====================================================================
// GEMM: C[M,N] = act(A[M,K] @ Bt[N,K]^T + bias)
//   A: f32 row-major, Bt: bf16 row-major [N,K] (column weights contiguous)
//   One wave computes a 16x64 tile with 4 independent accumulators and
//   explicit register double-buffering: next K-step's fragments are loaded
//   before the current step's 4 WMMAs issue, hiding the L2 latency.
//   Requires K % 32 == 0 and N % 64 == 0 (true for all uses here).
// =====================================================================
__global__ void k_gemm_bf(const float* __restrict__ A, const __bf16* __restrict__ Bt,
                          const float* __restrict__ bias, float* __restrict__ Cc,
                          int M, int Nn, int K, int relu) {
  int wid  = (blockIdx.x * blockDim.x + threadIdx.x) >> 5;
  int lane = threadIdx.x & 31;
  int tn_cnt = Nn >> 6;
  int tiles  = (M >> 4) * tn_cnt;
  if (wid >= tiles) return;
  int tm = wid / tn_cnt, tn = wid % tn_cnt;
  int half = lane >> 4, l15 = lane & 15;
  int arow = tm * 16 + l15;
  int bc0  = tn * 64 + l15;
  const float*  Ap  = A + (size_t)arow * K;
  const __bf16* Bp0 = Bt + (size_t)(bc0)      * K + half * 16;
  const __bf16* Bp1 = Bt + (size_t)(bc0 + 16) * K + half * 16;
  const __bf16* Bp2 = Bt + (size_t)(bc0 + 32) * K + half * 16;
  const __bf16* Bp3 = Bt + (size_t)(bc0 + 48) * K + half * 16;
  v8f c0 = {}, c1 = {}, c2 = {}, c3 = {};
  // prologue: fragments for k0 = 0
  v16bf a  = make_a_frag(Ap, half);
  v16bf b0 = *(const v16bf*)(Bp0);
  v16bf b1 = *(const v16bf*)(Bp1);
  v16bf b2 = *(const v16bf*)(Bp2);
  v16bf b3 = *(const v16bf*)(Bp3);
  for (int k0 = 32; k0 < K; k0 += 32) {
    // prefetch next K-step before consuming current fragments
    v16bf an  = make_a_frag(Ap + k0, half);
    v16bf bn0 = *(const v16bf*)(Bp0 + k0);
    v16bf bn1 = *(const v16bf*)(Bp1 + k0);
    v16bf bn2 = *(const v16bf*)(Bp2 + k0);
    v16bf bn3 = *(const v16bf*)(Bp3 + k0);
    c0 = __builtin_amdgcn_wmma_f32_16x16x32_bf16(false, a, false, b0, (short)0, c0, false, false);
    c1 = __builtin_amdgcn_wmma_f32_16x16x32_bf16(false, a, false, b1, (short)0, c1, false, false);
    c2 = __builtin_amdgcn_wmma_f32_16x16x32_bf16(false, a, false, b2, (short)0, c2, false, false);
    c3 = __builtin_amdgcn_wmma_f32_16x16x32_bf16(false, a, false, b3, (short)0, c3, false, false);
    a = an; b0 = bn0; b1 = bn1; b2 = bn2; b3 = bn3;
  }
  // epilogue
  c0 = __builtin_amdgcn_wmma_f32_16x16x32_bf16(false, a, false, b0, (short)0, c0, false, false);
  c1 = __builtin_amdgcn_wmma_f32_16x16x32_bf16(false, a, false, b1, (short)0, c1, false, false);
  c2 = __builtin_amdgcn_wmma_f32_16x16x32_bf16(false, a, false, b2, (short)0, c2, false, false);
  c3 = __builtin_amdgcn_wmma_f32_16x16x32_bf16(false, a, false, b3, (short)0, c3, false, false);

  float bv0 = bias ? bias[bc0]      : 0.0f;
  float bv1 = bias ? bias[bc0 + 16] : 0.0f;
  float bv2 = bias ? bias[bc0 + 32] : 0.0f;
  float bv3 = bias ? bias[bc0 + 48] : 0.0f;
#pragma unroll
  for (int v = 0; v < 8; ++v) {
    int row = tm * 16 + v + half * 8;
    float o0 = c0[v] + bv0, o1 = c1[v] + bv1, o2 = c2[v] + bv2, o3 = c3[v] + bv3;
    if (relu) {
      o0 = fmaxf(o0, 0.0f); o1 = fmaxf(o1, 0.0f);
      o2 = fmaxf(o2, 0.0f); o3 = fmaxf(o3, 0.0f);
    }
    float* cp = Cc + (size_t)row * Nn + bc0;
    cp[0] = o0; cp[16] = o1; cp[32] = o2; cp[48] = o3;
  }
}

// =====================================================================
// MHA scores: S[bh,i,j] = (Q . K) / sqrt(96). One wave = 16x64 tile.
// K loop is compile-time (96) -> fully unrolled by the compiler.
// =====================================================================
__global__ void k_attn_scores(const float* __restrict__ qkv, float* __restrict__ S) {
  int wid  = (blockIdx.x * blockDim.x + threadIdx.x) >> 5;
  int lane = threadIdx.x & 31;
  if (wid >= BGR * NH * 64) return;
  int bh = wid >> 6;
  int t  = wid & 63;
  int tm = t >> 2, tn = t & 3;
  int b = bh >> 2, h = bh & 3;
  int half = lane >> 4, l15 = lane & 15;
  const float* Q  = qkv + (size_t)(b * NPG) * (3 * HC) + h * CC;
  const float* Km = qkv + (size_t)(b * NPG) * (3 * HC) + HC + h * CC;
  int qrow = tm * 16 + l15;
  int kr0  = tn * 64 + l15;
  const float* Qp  = Q + (size_t)qrow * (3 * HC);
  const float* Kp0 = Km + (size_t)(kr0)      * (3 * HC);
  const float* Kp1 = Km + (size_t)(kr0 + 16) * (3 * HC);
  const float* Kp2 = Km + (size_t)(kr0 + 32) * (3 * HC);
  const float* Kp3 = Km + (size_t)(kr0 + 48) * (3 * HC);
  v8f c0 = {}, c1 = {}, c2 = {}, c3 = {};
#pragma unroll
  for (int k0 = 0; k0 < CC; k0 += 32) {
    v16bf a  = make_a_frag(Qp + k0, half);
    v16bf b0 = make_b_frag_f32(Kp0 + k0, half);
    v16bf b1 = make_b_frag_f32(Kp1 + k0, half);
    v16bf b2 = make_b_frag_f32(Kp2 + k0, half);
    v16bf b3 = make_b_frag_f32(Kp3 + k0, half);
    c0 = __builtin_amdgcn_wmma_f32_16x16x32_bf16(false, a, false, b0, (short)0, c0, false, false);
    c1 = __builtin_amdgcn_wmma_f32_16x16x32_bf16(false, a, false, b1, (short)0, c1, false, false);
    c2 = __builtin_amdgcn_wmma_f32_16x16x32_bf16(false, a, false, b2, (short)0, c2, false, false);
    c3 = __builtin_amdgcn_wmma_f32_16x16x32_bf16(false, a, false, b3, (short)0, c3, false, false);
  }
  const float alpha = 0.10206207261596575f; // 1/sqrt(96)
  float* Sp = S + (size_t)bh * NPG * NPG;
#pragma unroll
  for (int v = 0; v < 8; ++v) {
    int row = tm * 16 + v + half * 8;
    float* sp = Sp + (size_t)row * NPG + tn * 64 + l15;
    sp[0]  = c0[v] * alpha;
    sp[16] = c1[v] * alpha;
    sp[32] = c2[v] * alpha;
    sp[48] = c3[v] * alpha;
  }
}

// row softmax over 256 entries (mask is all-true)
__global__ void k_softmax256(float* __restrict__ S, int rows) {
  int r = blockIdx.x * blockDim.x + threadIdx.x;
  if (r >= rows) return;
  float* p = S + (size_t)r * NPG;
  float m = -1e30f;
  for (int j = 0; j < NPG; ++j) m = fmaxf(m, p[j]);
  float s = 0.0f;
  for (int j = 0; j < NPG; ++j) { float e = __expf(p[j] - m); p[j] = e; s += e; }
  float inv = 1.0f / s;
  for (int j = 0; j < NPG; ++j) p[j] *= inv;
}

// O[b*NPG+i, h*96+c] = sum_j P[bh,i,j] * V[b,j,h*96+c]
// B-side from pre-transposed bf16 Vt[bh,96,256] -> contiguous.
// One wave = 16x32 tile; 3 column tiles cover the 96-dim head.
__global__ void k_attn_apply(const float* __restrict__ S, const __bf16* __restrict__ Vt,
                             float* __restrict__ O) {
  int wid  = (blockIdx.x * blockDim.x + threadIdx.x) >> 5;
  int lane = threadIdx.x & 31;
  if (wid >= BGR * NH * 48) return;
  int bh = wid / 48;
  int t  = wid % 48;
  int tm = t / 3, tn = t % 3;
  int b = bh >> 2, h = bh & 3;
  int half = lane >> 4, l15 = lane & 15;
  const float* P = S + (size_t)bh * NPG * NPG;
  int prow  = tm * 16 + l15;
  int ocol0 = tn * 32 + l15;
  int ocol1 = ocol0 + 16;
  const float*  Pp  = P + (size_t)prow * NPG;
  const __bf16* Vp0 = Vt + ((size_t)bh * CC + ocol0) * NPG + half * 16;
  const __bf16* Vp1 = Vt + ((size_t)bh * CC + ocol1) * NPG + half * 16;
  v8f c0 = {}, c1 = {};
#pragma unroll
  for (int k0 = 0; k0 < NPG; k0 += 32) {
    v16bf a  = make_a_frag(Pp + k0, half);
    v16bf b0 = *(const v16bf*)(Vp0 + k0);
    v16bf b1 = *(const v16bf*)(Vp1 + k0);
    c0 = __builtin_amdgcn_wmma_f32_16x16x32_bf16(false, a, false, b0, (short)0, c0, false, false);
    c1 = __builtin_amdgcn_wmma_f32_16x16x32_bf16(false, a, false, b1, (short)0, c1, false, false);
  }
#pragma unroll
  for (int v = 0; v < 8; ++v) {
    int row = tm * 16 + v + half * 8;
    O[(size_t)(b * NPG + row) * HC + h * CC + ocol0] = c0[v];
    O[(size_t)(b * NPG + row) * HC + h * CC + ocol1] = c1[v];
  }
}

// =====================================================================
// Edge MLP: [emb(type), scalar] (18) -> 32 relu -> 16
// =====================================================================
__global__ void k_edge_mlp(const int* __restrict__ etype, const float* __restrict__ escalar,
                           const float* __restrict__ eemb, const float* __restrict__ W1,
                           const float* __restrict__ b1, const float* __restrict__ W2,
                           const float* __restrict__ b2, float* __restrict__ ea) {
  int e = blockIdx.x * blockDim.x + threadIdx.x;
  if (e >= N_EDGES) return;
  float in[18];
  int t = etype[e];
#pragma unroll
  for (int k = 0; k < 6; ++k) in[k] = eemb[t * 6 + k];
#pragma unroll
  for (int k = 0; k < 12; ++k) in[6 + k] = escalar[(size_t)e * 12 + k];
  float h[32];
  for (int j = 0; j < 32; ++j) {
    float acc = b1[j];
#pragma unroll
    for (int k = 0; k < 18; ++k) acc += in[k] * W1[k * 32 + j];
    h[j] = fmaxf(acc, 0.0f);
  }
  for (int j = 0; j < 16; ++j) {
    float acc = b2[j];
#pragma unroll
    for (int k = 0; k < 32; ++k) acc += h[k] * W2[k * 16 + j];
    ea[(size_t)e * 16 + j] = acc;
  }
}

// =====================================================================
// GATv2 edge kernels (one thread per (edge, head))
// =====================================================================
__global__ void k_gat_edge(const float* __restrict__ xl, const float* __restrict__ xr,
                           const float* __restrict__ ea, const float* __restrict__ We,
                           const float* __restrict__ att, const int* __restrict__ src,
                           const int* __restrict__ dst, float* __restrict__ logit,
                           unsigned* __restrict__ mx) {
  int tid = blockIdx.x * blockDim.x + threadIdx.x;
  if (tid >= N_EDGES * NH) return;
  int e = tid >> 2, h = tid & 3;
  int s = src[e], d = dst[e];
  float ev[16];
#pragma unroll
  for (int k = 0; k < 16; ++k) ev[k] = ea[(size_t)e * 16 + k];
  const float* xlp = xl + (size_t)s * HC + h * CC;
  const float* xrp = xr + (size_t)d * HC + h * CC;
  const float* ap  = att + h * CC;
  const float* Wp  = We + h * CC;
  float lg = 0.0f;
  for (int c = 0; c < CC; ++c) {
    float m = xlp[c] + xrp[c];
#pragma unroll
    for (int k = 0; k < 16; ++k) m += ev[k] * Wp[(size_t)k * HC + c];
    m = (m > 0.0f) ? m : 0.2f * m;   // leaky_relu(0.2)
    lg += m * ap[c];
  }
  logit[tid] = lg;
  atomicMax(&mx[d * NH + h], f2ord(lg));
}

__global__ void k_gat_expsum(const float* __restrict__ logit, const unsigned* __restrict__ mx,
                             const int* __restrict__ dst, float* __restrict__ aatt,
                             float* __restrict__ den) {
  int tid = blockIdx.x * blockDim.x + threadIdx.x;
  if (tid >= N_EDGES * NH) return;
  int e = tid >> 2, h = tid & 3;
  int d = dst[e];
  float m = ord2f(mx[d * NH + h]);
  float av = __expf(logit[tid] - m);
  aatt[tid] = av;
  atomicAdd(&den[d * NH + h], av);
}

__global__ void k_gat_agg(const float* __restrict__ aatt, const float* __restrict__ den,
                          const float* __restrict__ xl, const int* __restrict__ src,
                          const int* __restrict__ dst, float* __restrict__ out) {
  int tid = blockIdx.x * blockDim.x + threadIdx.x;
  if (tid >= N_EDGES * NH) return;
  int e = tid >> 2, h = tid & 3;
  int s = src[e], d = dst[e];
  float w = aatt[tid] / den[d * NH + h];
  const float* xlp = xl + (size_t)s * HC + h * CC;
  float* op = out + (size_t)d * HC + h * CC;
  for (int c = 0; c < CC; ++c) atomicAdd(&op[c], xlp[c] * w);
}

// =====================================================================
// small elementwise / reduction kernels
// =====================================================================
__global__ void k_zero_f32(float* p, int n) {
  int i = blockIdx.x * blockDim.x + threadIdx.x;
  if (i < n) p[i] = 0.0f;
}
__global__ void k_zero_u32(unsigned* p, int n) {
  int i = blockIdx.x * blockDim.x + threadIdx.x;
  if (i < n) p[i] = 0u;
}
__global__ void k_fill_bias(float* __restrict__ out, const float* __restrict__ bias, int n, int ld) {
  int i = blockIdx.x * blockDim.x + threadIdx.x;
  if (i < n) out[i] = bias[i % ld];
}
// in-place relu then BN (eval)
__global__ void k_bn_relu(float* __restrict__ x, const float* __restrict__ g,
                          const float* __restrict__ b, const float* __restrict__ m,
                          const float* __restrict__ v, int n) {
  int i = blockIdx.x * blockDim.x + threadIdx.x;
  if (i >= n) return;
  int c = i % HC;
  float xv = fmaxf(x[i], 0.0f);
  x[i] = (xv - m[c]) * rsqrtf(v[c] + EPS_BN) * g[c] + b[c];
}
// per-graph mean pool over the contiguous 256 nodes of each graph
__global__ void k_pool_mean(const float* __restrict__ X, float* __restrict__ G) {
  int tid = blockIdx.x * blockDim.x + threadIdx.x;
  if (tid >= BGR * HC) return;
  int b = tid / HC, c = tid % HC;
  float s = 0.0f;
  for (int i = 0; i < NPG; ++i) s += X[(size_t)(b * NPG + i) * HC + c];
  G[tid] = s * (1.0f / NPG);
}

// =====================================================================
// SeqCNN: conv1d(k=3,pad=1) & conv1d(k=5,pad=2) + relu + max over L
// =====================================================================
__global__ void k_seq_conv(const int* __restrict__ seq_ids, const float* __restrict__ emb,
                           const float* __restrict__ W3, const float* __restrict__ b3,
                           const float* __restrict__ W5, const float* __restrict__ b5,
                           float* __restrict__ p3, float* __restrict__ p5) {
  __shared__ float r3[256], r5[256];
  int b = blockIdx.x / SEQCH, oc = blockIdx.x % SEQCH;
  int l = threadIdx.x;
  const int* sb = seq_ids + b * NPG;
  float a3 = b3[oc], a5 = b5[oc];
  for (int t = 0; t < 3; ++t) {
    int pos = l + t - 1;
    if (pos >= 0 && pos < NPG) {
      const float* ep = emb + sb[pos] * EMBD;
#pragma unroll
      for (int ic = 0; ic < EMBD; ++ic) a3 += W3[(oc * EMBD + ic) * 3 + t] * ep[ic];
    }
  }
  for (int t = 0; t < 5; ++t) {
    int pos = l + t - 2;
    if (pos >= 0 && pos < NPG) {
      const float* ep = emb + sb[pos] * EMBD;
#pragma unroll
      for (int ic = 0; ic < EMBD; ++ic) a5 += W5[(oc * EMBD + ic) * 5 + t] * ep[ic];
    }
  }
  r3[l] = fmaxf(a3, 0.0f);
  r5[l] = fmaxf(a5, 0.0f);
  __syncthreads();
  for (int s = 128; s > 0; s >>= 1) {
    if (l < s) {
      r3[l] = fmaxf(r3[l], r3[l + s]);
      r5[l] = fmaxf(r5[l], r5[l + s]);
    }
    __syncthreads();
  }
  if (l == 0) { p3[b * SEQCH + oc] = r3[0]; p5[b * SEQCH + oc] = r5[0]; }
}

__global__ void k_seq_proj(const float* __restrict__ p3, const float* __restrict__ p5,
                           const float* __restrict__ Wp, const float* __restrict__ bp,
                           float* __restrict__ semb) {
  int tid = blockIdx.x * blockDim.x + threadIdx.x;
  if (tid >= BGR * SEQCH) return;
  int b = tid / SEQCH, oc = tid % SEQCH;
  float acc = bp[oc];
#pragma unroll
  for (int k = 0; k < SEQCH; ++k) acc += p3[b * SEQCH + k] * Wp[k * SEQCH + oc];
#pragma unroll
  for (int k = 0; k < SEQCH; ++k) acc += p5[b * SEQCH + k] * Wp[(SEQCH + k) * SEQCH + oc];
  semb[tid] = fmaxf(acc, 0.0f);
}

// head: fused(816) -> 96 relu -> 1 ; one block per graph
__global__ void k_head(const float* __restrict__ g_emb, const float* __restrict__ semb,
                       const float* __restrict__ g_glob, const float* __restrict__ W1,
                       const float* __restrict__ b1, const float* __restrict__ W2,
                       const float* __restrict__ b2, float* __restrict__ out) {
  __shared__ float h1[96];
  int b = blockIdx.x, j = threadIdx.x;
  if (j < 96) {
    float acc = b1[j];
    for (int k = 0; k < HC; ++k) acc += g_emb[b * HC + k] * W1[k * 96 + j];
    for (int k = 0; k < SEQCH; ++k) acc += semb[b * SEQCH + k] * W1[(HC + k) * 96 + j];
    for (int k = 0; k < HC; ++k) acc += g_glob[b * HC + k] * W1[(HC + SEQCH + k) * 96 + j];
    h1[j] = fmaxf(acc, 0.0f);
  }
  __syncthreads();
  if (j == 0) {
    float s = b2[0];
    for (int k = 0; k < 96; ++k) s += h1[k] * W2[k];
    out[b] = s;
  }
}

// =====================================================================
// host side
// =====================================================================
static inline int cdiv(long a, long b) { return (int)((a + b - 1) / b); }

extern "C" void kernel_launch(void* const* d_in, const int* in_sizes, int n_in,
                              void* d_out, int out_size, void* d_ws, size_t ws_size,
                              hipStream_t stream) {
  // ------- resolve flattened-input indices (insertion order vs jax-sorted) -------
  int iX, iES, iEE, iW1, ib1, iW2, ib2;
  int cWl[3], cbl[3], cWr[3], cbr[3], cWe[3], cAtt[3], cBias[3];
  int bg[3], bb[3], bm[3], bv[3];
  int iWin, ibin, iWout, ibout;
  int iSemb, iW3, ib3, iW5, ib5, iWp, ibp;
  int ihW1, ihb1, ihW2, ihb2;
  int iEI, iET, iBatch, iSeq;
  bool insertion = (n_in > 2 && in_sizes[0] == N_NODES * INF_D);
  if (insertion) {
    iX = 0; iES = 1; iEE = 2; iW1 = 3; ib1 = 4; iW2 = 5; ib2 = 6;
    for (int l = 0; l < 3; ++l) {
      int b0 = 7 + 7 * l;
      cWl[l] = b0; cbl[l] = b0 + 1; cWr[l] = b0 + 2; cbr[l] = b0 + 3;
      cWe[l] = b0 + 4; cAtt[l] = b0 + 5; cBias[l] = b0 + 6;
    }
    for (int l = 0; l < 3; ++l) {
      int b0 = 28 + 4 * l;
      bg[l] = b0; bb[l] = b0 + 1; bm[l] = b0 + 2; bv[l] = b0 + 3;
    }
    iWin = 40; ibin = 41; iWout = 42; ibout = 43;
    iSemb = 44; iW3 = 45; ib3 = 46; iW5 = 47; ib5 = 48; iWp = 49; ibp = 50;
    ihW1 = 51; ihb1 = 52; ihW2 = 53; ihb2 = 54;
    iEI = 55; iET = 56; iBatch = 57; iSeq = 58;
  } else { // jax tree_flatten: dict keys sorted at every level
    iBatch = 0; iEI = 1; iES = 2; iET = 3;
    for (int l = 0; l < 3; ++l) {
      int b0 = 4 + 4 * l;
      bb[l] = b0; bg[l] = b0 + 1; bm[l] = b0 + 2; bv[l] = b0 + 3;
    }
    for (int l = 0; l < 3; ++l) {
      int b0 = 16 + 7 * l;
      cWe[l] = b0; cWl[l] = b0 + 1; cWr[l] = b0 + 2; cAtt[l] = b0 + 3;
      cBias[l] = b0 + 4; cbl[l] = b0 + 5; cbr[l] = b0 + 6;
    }
    iEE = 37; iW1 = 38; iW2 = 39; ib1 = 40; ib2 = 41;
    ihW1 = 42; ihW2 = 43; ihb1 = 44; ihb2 = 45;
    iWin = 46; iWout = 47; ibin = 48; ibout = 49;
    iW3 = 50; iW5 = 51; iWp = 52; ib3 = 53; ib5 = 54; ibp = 55; iSemb = 56;
    iSeq = 57; iX = 58;
  }
  (void)iBatch;

  const float* x_in   = (const float*)d_in[iX];
  const float* esc    = (const float*)d_in[iES];
  const int*   e_src  = (const int*)d_in[iEI];
  const int*   e_dst  = e_src + N_EDGES;
  const int*   e_type = (const int*)d_in[iET];
  const int*   seq    = (const int*)d_in[iSeq];
  float* out = (float*)d_out;

  // ------- workspace map (float units; bf16 regions carved as float-sized) -------
  float* ws = (float*)d_ws;
  size_t off = 0;
  float* x0    = ws + off; off += (size_t)N_NODES * HC;
  float* x1    = ws + off; off += (size_t)N_NODES * HC;
  float* xl    = ws + off; off += (size_t)N_NODES * HC;   // also reused as attn O
  float* xr    = ws + off; off += (size_t)N_NODES * HC;   // also reused as attn_out
  float* ea    = ws + off; off += (size_t)N_EDGES * 16;
  float* logit = ws + off; off += (size_t)N_EDGES * NH;
  float* aatt  = ws + off; off += (size_t)N_EDGES * NH;
  unsigned* mx = (unsigned*)(ws + off); off += (size_t)N_NODES * NH;
  float* den   = ws + off; off += (size_t)N_NODES * NH;
  float* qkv   = ws + off; off += (size_t)N_NODES * 3 * HC;
  float* S     = ws + off; off += (size_t)BGR * NH * NPG * NPG;
  float* g_emb = ws + off; off += (size_t)BGR * HC;
  float* g_glo = ws + off; off += (size_t)BGR * HC;
  float* p3    = ws + off; off += (size_t)BGR * SEQCH;
  float* p5    = ws + off; off += (size_t)BGR * SEQCH;
  float* semb  = ws + off; off += (size_t)BGR * SEQCH;
  // bf16 weight staging
  __bf16* wlt   = (__bf16*)(ws + off); off += (size_t)HC * HC / 2;        // [384, K<=384]
  __bf16* wrt   = (__bf16*)(ws + off); off += (size_t)HC * HC / 2;
  __bf16* wint  = (__bf16*)(ws + off); off += (size_t)3 * HC * HC / 2;    // [1152,384]
  __bf16* woutt = (__bf16*)(ws + off); off += (size_t)HC * HC / 2;        // [384,384]
  __bf16* vt    = (__bf16*)(ws + off); off += (size_t)BGR * NH * CC * NPG / 2;
  (void)ws_size;

  const int TB = 256;
  auto gemm_blocks = [](int M, int N) { return cdiv((long)(M / 16) * (N / 64) * 32, 256); };

  // ------- prepare torch-style bf16 weights for the MHA projections -------
  k_f32_to_bf<<<cdiv(3 * HC * HC, TB), TB, 0, stream>>>((const float*)d_in[iWin], wint, 3 * HC * HC);
  k_f32_to_bf<<<cdiv(HC * HC, TB), TB, 0, stream>>>((const float*)d_in[iWout], woutt, HC * HC);

  // ------- edge feature MLP -------
  k_edge_mlp<<<cdiv(N_EDGES, TB), TB, 0, stream>>>(
      e_type, esc, (const float*)d_in[iEE], (const float*)d_in[iW1],
      (const float*)d_in[ib1], (const float*)d_in[iW2], (const float*)d_in[ib2], ea);

  // ------- 3 GATv2 layers -------
  const float* xcur = x_in;
  float* agg[2] = {x0, x1};
  for (int l = 0; l < 3; ++l) {
    int K = (l == 0) ? INF_D : HC;
    float* aggbuf = agg[l & 1];
    // transpose + bf16 the jax-style [K,384] weights
    k_w_nn_to_bf_t<<<cdiv(K * HC, TB), TB, 0, stream>>>((const float*)d_in[cWl[l]], wlt, K, HC);
    k_w_nn_to_bf_t<<<cdiv(K * HC, TB), TB, 0, stream>>>((const float*)d_in[cWr[l]], wrt, K, HC);
    k_gemm_bf<<<gemm_blocks(N_NODES, HC), TB, 0, stream>>>(
        xcur, wlt, (const float*)d_in[cbl[l]], xl, N_NODES, HC, K, 0);
    k_gemm_bf<<<gemm_blocks(N_NODES, HC), TB, 0, stream>>>(
        xcur, wrt, (const float*)d_in[cbr[l]], xr, N_NODES, HC, K, 0);
    k_fill_bias<<<cdiv((long)N_NODES * HC, TB), TB, 0, stream>>>(
        aggbuf, (const float*)d_in[cBias[l]], N_NODES * HC, HC);
    k_zero_u32<<<cdiv(N_NODES * NH, TB), TB, 0, stream>>>(mx, N_NODES * NH);
    k_zero_f32<<<cdiv(N_NODES * NH, TB), TB, 0, stream>>>(den, N_NODES * NH);
    k_gat_edge<<<cdiv(N_EDGES * NH, TB), TB, 0, stream>>>(
        xl, xr, ea, (const float*)d_in[cWe[l]], (const float*)d_in[cAtt[l]],
        e_src, e_dst, logit, mx);
    k_gat_expsum<<<cdiv(N_EDGES * NH, TB), TB, 0, stream>>>(logit, mx, e_dst, aatt, den);
    k_gat_agg<<<cdiv(N_EDGES * NH, TB), TB, 0, stream>>>(aatt, den, xl, e_src, e_dst, aggbuf);
    k_bn_relu<<<cdiv((long)N_NODES * HC, TB), TB, 0, stream>>>(
        aggbuf, (const float*)d_in[bg[l]], (const float*)d_in[bb[l]],
        (const float*)d_in[bm[l]], (const float*)d_in[bv[l]], N_NODES * HC);
    xcur = aggbuf;
  }
  const float* xfin = xcur;

  // ------- graph mean pool -------
  k_pool_mean<<<cdiv(BGR * HC, TB), TB, 0, stream>>>(xfin, g_emb);

  // ------- MHA over each graph -------
  k_gemm_bf<<<gemm_blocks(N_NODES, 3 * HC), TB, 0, stream>>>(
      xfin, wint, (const float*)d_in[ibin], qkv, N_NODES, 3 * HC, HC, 0);
  k_attn_scores<<<cdiv((long)BGR * NH * 64 * 32, TB), TB, 0, stream>>>(qkv, S);
  k_softmax256<<<cdiv(BGR * NH * NPG, TB), TB, 0, stream>>>(S, BGR * NH * NPG);
  k_make_vt<<<cdiv(BGR * NH * CC * NPG, TB), TB, 0, stream>>>(qkv, vt);
  k_attn_apply<<<cdiv((long)BGR * NH * 48 * 32, TB), TB, 0, stream>>>(S, vt, xl);
  k_gemm_bf<<<gemm_blocks(N_NODES, HC), TB, 0, stream>>>(
      xl, woutt, (const float*)d_in[ibout], xr, N_NODES, HC, HC, 0);
  k_pool_mean<<<cdiv(BGR * HC, TB), TB, 0, stream>>>(xr, g_glo);

  // ------- SeqCNN branch -------
  k_seq_conv<<<BGR * SEQCH, 256, 0, stream>>>(
      seq, (const float*)d_in[iSemb], (const float*)d_in[iW3], (const float*)d_in[ib3],
      (const float*)d_in[iW5], (const float*)d_in[ib5], p3, p5);
  k_seq_proj<<<cdiv(BGR * SEQCH, TB), TB, 0, stream>>>(
      p3, p5, (const float*)d_in[iWp], (const float*)d_in[ibp], semb);

  // ------- head -------
  k_head<<<BGR, 128, 0, stream>>>(
      g_emb, semb, g_glo, (const float*)d_in[ihW1], (const float*)d_in[ihb1],
      (const float*)d_in[ihW2], (const float*)d_in[ihb2], out);
  (void)out_size;
}
